// QuantLinear_75196287418853
// MI455X (gfx1250) — 8-bit matrix operand order repaired and checked by reading, compiled, not run on hardware
//
#include <hip/hip_runtime.h>
#include <hip/hip_bf16.h>


typedef __attribute__((ext_vector_type(16))) int   v16i;
typedef __attribute__((ext_vector_type(8)))  float v8f;

#define M_DIM 8192
#define N_DIM 8192
#define K_DIM 2048
#define KT    (K_DIM / 128)
#define TILE_BYTES 2048
#define E4M3_MAX 448.0f

#if __has_builtin(__builtin_amdgcn_cvt_pk_fp8_f32) && __has_builtin(__builtin_amdgcn_cvt_f32_fp8)
#define QL_HW_FP8_CVT 1
#else
#define QL_HW_FP8_CVT 0
#endif

__device__ __forceinline__ unsigned ql_enc_e4m3_sw(float f) {
  float cf = fminf(fmaxf(f, -E4M3_MAX), E4M3_MAX);
  unsigned s = (__float_as_uint(cf) >> 31) & 1u;
  float af = fabsf(cf);
  unsigned bits;
  if (af < 0.015625f) {
    bits = (unsigned)rintf(af * 512.0f);
  } else {
    int e; float m = frexpf(af, &e);
    unsigned q = (unsigned)rintf(m * 16.0f);
    int ee = e - 1;
    if (q == 16) { q = 8; ee += 1; }
    bits = ((unsigned)(ee + 7) << 3) | (q - 8u);
  }
  return bits | (s << 7);
}

__device__ __forceinline__ float ql_dec_e4m3_sw(unsigned v) {
  unsigned s = (v >> 7) & 1u, e = (v >> 3) & 0xFu, m = v & 7u;
  float r = e ? ldexpf((float)(8u + m), (int)e - 10) : ldexpf((float)m, -9);
  return s ? -r : r;
}

__device__ __forceinline__ float ql_qdq_e4m3(float f) {
  f = fminf(fmaxf(f, -E4M3_MAX), E4M3_MAX);
#if QL_HW_FP8_CVT
  int pk = __builtin_amdgcn_cvt_pk_fp8_f32(f, f, 0, false);
  return __builtin_amdgcn_cvt_f32_fp8(pk, 0);
#else
  return ql_dec_e4m3_sw(ql_enc_e4m3_sw(f));
#endif
}

__device__ __forceinline__ unsigned ql_pack4_e4m3(float4 v) {
  float c0 = fminf(fmaxf(v.x, -E4M3_MAX), E4M3_MAX);
  float c1 = fminf(fmaxf(v.y, -E4M3_MAX), E4M3_MAX);
  float c2 = fminf(fmaxf(v.z, -E4M3_MAX), E4M3_MAX);
  float c3 = fminf(fmaxf(v.w, -E4M3_MAX), E4M3_MAX);
#if QL_HW_FP8_CVT
  int pk = __builtin_amdgcn_cvt_pk_fp8_f32(c0, c1, 0, false);
  pk     = __builtin_amdgcn_cvt_pk_fp8_f32(c2, c3, pk, true);
  return (unsigned)pk;
#else
  return ql_enc_e4m3_sw(c0)         | (ql_enc_e4m3_sw(c1) << 8) |
         (ql_enc_e4m3_sw(c2) << 16) | (ql_enc_e4m3_sw(c3) << 24);
#endif
}

__device__ __forceinline__ float ql_bf16_rne(float f) {
  unsigned u = __float_as_uint(f);
  unsigned r = (u + 0x7FFFu + ((u >> 16) & 1u)) & 0xFFFF0000u;
  return __uint_as_float(r);
}

__global__ __launch_bounds__(256)
void ql_quant_swizzle_a(const float* __restrict__ in, unsigned* __restrict__ out) {
  unsigned i = blockIdx.x * blockDim.x + threadIdx.x;
  unsigned o    = i << 2;
  unsigned tile = o >> 11;
  unsigned oin  = o & 2047u;
  unsigned row_tile = tile >> 4;
  unsigned k_tile   = tile & (KT - 1u);
  unsigned c   = oin >> 9;
  unsigned rem = oin & 511u;
  unsigned L   = rem >> 4;
  unsigned bo  = rem & 15u;
  unsigned v   = c * 4u + (bo >> 2);
  unsigned hi  = L >> 4;
  unsigned k   = k_tile * 128u + (v >> 3) * 64u + ((v & 7u) >> 1) * 16u
               + hi * 8u + (v & 1u) * 4u;
  unsigned r   = row_tile * 16u + (L & 15u);
  float4 f = *(const float4*)(in + (size_t)r * K_DIM + k);
  out[i] = ql_pack4_e4m3(f);
}

__global__ __launch_bounds__(256)
void ql_quant_swizzle_b(const float* __restrict__ in, unsigned* __restrict__ out) {
  unsigned i = blockIdx.x * blockDim.x + threadIdx.x;
  unsigned o    = i << 2;
  unsigned tile = o >> 11;
  unsigned oin  = o & 2047u;
  unsigned col_tile = tile >> 4;
  unsigned k_tile   = tile & (KT - 1u);
  unsigned c   = oin >> 9;
  unsigned rem = oin & 511u;
  unsigned L   = rem >> 4;
  unsigned bo  = rem & 15u;
  unsigned n   = col_tile * 16u + (L & 15u);
  unsigned k   = k_tile * 128u + (c >> 1) * 64u + ((c & 1u) * 2u + (bo >> 3)) * 16u + (L >> 4) * 8u + (bo & 7u);
  float4 f = *(const float4*)(in + (size_t)n * K_DIM + k);
  out[i] = ql_pack4_e4m3(f);
}

__global__ __launch_bounds__(256)
void ql_gemm_fp8(const unsigned char* __restrict__ Aq,
                 const unsigned char* __restrict__ Bq,
                 const float* __restrict__ bias,
                 float* __restrict__ out) {
  const int lane = threadIdx.x & 31;
  const int wave = threadIdx.x >> 5;
  const int wm   = wave & 1;
  const int wn   = wave >> 1;
  const int m0   = blockIdx.y * 128 + wm * 64;
  const int n0   = blockIdx.x * 256 + wn * 64;
  const int hi   = lane >> 4;
  const int ln   = lane & 15;

  v8f acc[4][4] = {};

  const unsigned char* aBase = Aq + (size_t)(m0 >> 4) * KT * TILE_BYTES + lane * 16;
  const unsigned char* bBase = Bq + (size_t)(n0 >> 4) * KT * TILE_BYTES + lane * 16;

  for (int kt = 0; kt < KT; ++kt) {
    v16i a[4], b[4];
    #pragma unroll
    for (int t = 0; t < 4; ++t) {
      #pragma unroll
      for (int c = 0; c < 4; ++c) {
        int4 d = *(const int4*)(aBase + ((size_t)t * KT + kt) * TILE_BYTES + c * 512);
        a[t][4 * c] = d.x; a[t][4 * c + 1] = d.y; a[t][4 * c + 2] = d.z; a[t][4 * c + 3] = d.w;
      }
    }
    #pragma unroll
    for (int u = 0; u < 4; ++u) {
      #pragma unroll
      for (int c = 0; c < 4; ++c) {
        int4 d = *(const int4*)(bBase + ((size_t)u * KT + kt) * TILE_BYTES + c * 512);
        b[u][4 * c] = d.x; b[u][4 * c + 1] = d.y; b[u][4 * c + 2] = d.z; b[u][4 * c + 3] = d.w;
      }
    }
    #pragma unroll
    for (int t = 0; t < 4; ++t) {
      #pragma unroll
      for (int u = 0; u < 4; ++u) {
        acc[t][u] = __builtin_amdgcn_wmma_f32_16x16x128_fp8_fp8(
            a[t], b[u], (short)0, acc[t][u], false, false);
      }
    }
  }

  #pragma unroll
  for (int u = 0; u < 4; ++u) {
    const int gn = n0 + u * 16 + ln;
    const float bq = ql_qdq_e4m3(bias[gn]);
    #pragma unroll
    for (int t = 0; t < 4; ++t) {
      const int gmBase = m0 + t * 16 + hi * 8;
      #pragma unroll
      for (int r = 0; r < 8; ++r) {
        float v = acc[t][u][r] + bq;
        out[(size_t)(gmBase + r) * N_DIM + gn] = ql_bf16_rne(v);
      }
    }
  }
}

extern "C" void kernel_launch(void* const* d_in, const int* in_sizes, int n_in,
                              void* d_out, int out_size, void* d_ws, size_t ws_size,
                              hipStream_t stream) {
  const float* x = (const float*)d_in[0];
  const float* w = (const float*)d_in[1];
  const float* b = (const float*)d_in[2];
  float* out = (float*)d_out;

  unsigned char* xq = (unsigned char*)d_ws;
  unsigned char* wq = xq + (size_t)M_DIM * K_DIM;

  const int n4 = (M_DIM * K_DIM) / 4;
  ql_quant_swizzle_a<<<n4 / 256, 256, 0, stream>>>(x, (unsigned*)xq);
  ql_quant_swizzle_b<<<n4 / 256, 256, 0, stream>>>(w, (unsigned*)wq);

  dim3 grid(N_DIM / 256, M_DIM / 128);
  ql_gemm_fp8<<<grid, dim3(256), 0, stream>>>(xq, wq, b, out);
}
